// GANOKernel_12163347383246
// MI455X (gfx1250) — matrix operand order repaired and checked by reading, not compiled, not run on hardware
//
#include <hip/hip_runtime.h>
#include <math.h>

// ---------------------------------------------------------------------------
// GANO geometric attention for gfx1250 (MI455X).
//   hidden(q,o) = relu( Aq[q,:] + Ao[o,:] + dist(q,o)*W1[9,:] )   (factorized)
//   logits GEMM : hidden @ W2       -> V_WMMA_F32_16X16X32_BF16 (K=32/instr)
//   P @ V       : bf16 in, f32 acc  -> V_WMMA_F32_16X16X32_BF16 (K=32/instr)
//   V = h_obs@Wv: one-shot f32 WMMA -> V_WMMA_F32_16X16X4_F32 (stores bf16)
//   HBM->LDS    : double-buffered tensor_load_to_lds (TDM), s_wait_tensorcnt 2
//                 overlaps DMA of chunk i+1 with compute of chunk i.
// ---------------------------------------------------------------------------

typedef __attribute__((ext_vector_type(2)))  float        v2f;
typedef __attribute__((ext_vector_type(4)))  float        v4f;
typedef __attribute__((ext_vector_type(8)))  float        v8f;
typedef __attribute__((ext_vector_type(16))) __bf16       v16bf;
typedef __attribute__((ext_vector_type(4)))  unsigned int u32x4;
typedef __attribute__((ext_vector_type(8)))  int          i32x8;
typedef __attribute__((ext_vector_type(4)))  int          i32x4;

#define WMMA_F32X4(a, b, c) \
  __builtin_amdgcn_wmma_f32_16x16x4_f32(false, (a), false, (b), (short)0, (c), false, false)
#define WMMA_BF16(a, b, c) \
  __builtin_amdgcn_wmma_f32_16x16x32_bf16(false, (a), false, (b), (short)0, (c), false, false)

#if __has_builtin(__builtin_amdgcn_tensor_load_to_lds) && \
    __has_builtin(__builtin_amdgcn_s_wait_tensorcnt)
#define GANO_HAVE_TDM 1
#else
#define GANO_HAVE_TDM 0
#endif

constexpr int   B_  = 2;
constexpr int   NQ_ = 1024;
constexpr int   NO_ = 1024;
constexpr int   D_  = 64;
constexpr int   H_  = 4;
constexpr int   CH_ = 32;                    // obs per chunk
constexpr int   NCHUNK_ = NO_ / CH_;         // 32
constexpr float RADIUS_ = 2.0f;

#if GANO_HAVE_TDM
// Issue one 2D-tile TDM load (group2/3 NULL-equivalent zeros; cpol 0).
__device__ __forceinline__ void gano_tdm_load(unsigned ldsOff, const void* gptr,
                                              i32x8 g1) {
  unsigned long long ga = (unsigned long long)gptr;
  u32x4 g0;
  g0[0] = 1u;                                             // count = 1
  g0[1] = ldsOff;                                         // lds_addr
  g0[2] = (unsigned)ga;                                   // global_addr lo32
  g0[3] = (unsigned)((ga >> 32) & 0x01FFFFFFu) | 0x80000000u;  // hi | type=2
  i32x4 z4 = {0, 0, 0, 0};
#if __clang_major__ >= 23
  i32x8 z8 = {0, 0, 0, 0, 0, 0, 0, 0};
  __builtin_amdgcn_tensor_load_to_lds(g0, g1, z4, z4, z8, 0);
#else
  __builtin_amdgcn_tensor_load_to_lds(g0, g1, z4, z4, 0);
#endif
}

// D# group 1 for a CH_ x 64 tile with row stride 64, element size szCode
// (2 = 4B f32, 1 = 2B bf16).
__device__ __forceinline__ i32x8 gano_tdm_g1(unsigned szCode) {
  i32x8 g1;
  g1[0] = (int)(szCode << 16);               // data_size
  g1[1] = (int)(64u << 16);                  // tensor_dim0 = 64   [79:48]
  g1[2] = (int)(((unsigned)NO_) << 16);      // tensor_dim1 = No   [111:80]
  g1[3] = (int)(64u << 16);                  // tile_dim0 = 64     [127:112]
  g1[4] = CH_;                               // tile_dim1 = 32     [143:128]
  g1[5] = 64;                                // tensor_dim0_stride = 64
  g1[6] = 0;
  g1[7] = 0;
  return g1;
}
#endif

// --------------------------- stage 1a: Aq / Ao ------------------------------
__global__ void gano_precompute_rows(const float* __restrict__ pos,
                                     const float* __restrict__ W1,
                                     const float* __restrict__ b1,
                                     float* __restrict__ out,
                                     int n, int rowA, int rowB, float signB, int addB1) {
  int idx = blockIdx.x * blockDim.x + threadIdx.x;
  if (idx >= n) return;
  int i = idx >> 6, d = idx & 63;
  float acc = addB1 ? b1[d] : 0.0f;
#pragma unroll
  for (int p = 0; p < 3; ++p) {
    float w = W1[(rowA + p) * 64 + d] + signB * W1[(rowB + p) * 64 + d];
    acc += pos[i * 3 + p] * w;
  }
  out[idx] = acc;
}

// ----------------- stage 1b: V = h_obs@Wv + bv (f32 WMMA, bf16 store) -------
__global__ void gano_value_proj(const float* __restrict__ h_obs,
                                const float* __restrict__ Wv,
                                const float* __restrict__ bv,
                                __bf16* __restrict__ V) {
  int lane = threadIdx.x;
  int rt = blockIdx.x >> 2;
  int ct = blockIdx.x & 3;
  int half = lane >> 4;
  int mrow = lane & 15;
  int ncol = ct * 16 + (lane & 15);

  v8f c;
  float bvn = bv[ncol];
#pragma unroll
  for (int r = 0; r < 8; ++r) c[r] = bvn;

  const float* arow = h_obs + (rt * 16 + mrow) * 64;
#pragma unroll
  for (int g = 0; g < 16; ++g) {
    int k0 = 4 * g + 2 * half;
    v2f a, b;
    a.x = arow[k0];
    a.y = arow[k0 + 1];
    b.x = Wv[k0 * 64 + ncol];
    b.y = Wv[(k0 + 1) * 64 + ncol];
    c = WMMA_F32X4(a, b, c);
  }
#pragma unroll
  for (int r = 0; r < 8; ++r) {
    int M = r + half * 8;
    V[(rt * 16 + M) * 64 + ncol] = (__bf16)c[r];
  }
}

// --------------------------- stage 2: attention -----------------------------
// grid = B * (Nq/16) blocks, 32 threads (one wave) per block.
__global__ void gano_attention(const float* __restrict__ pos_q_g,
                               const float* __restrict__ pos_o_g,
                               const int*   __restrict__ omask_g,
                               const float* __restrict__ W1,
                               const float* __restrict__ W2,
                               const float* __restrict__ b2,
                               const float* __restrict__ Aq_g,
                               const float* __restrict__ Ao_g,
                               const __bf16* __restrict__ V_g,
                               float* __restrict__ out) {
  __shared__ __align__(16) float  sAo[2][CH_ * 64];   // TDM dst, f32
  __shared__ __align__(16) __bf16 sV[2][CH_ * 64];    // TDM dst, bf16
  __shared__ float sDist[CH_][16];                    // [o][q]
  __shared__ float sLog[H_][CH_][16];                 // logits -> probs [h][o][q]
  __shared__ float sScale[H_][16];
  __shared__ float sS[H_][16];
  __shared__ float sQpos[16][4];
  __shared__ float sOpos[CH_][4];
  __shared__ int   sMask[CH_];

  const int lane = threadIdx.x;
  const int half = lane >> 4;
  const int lq   = lane & 15;
  const int bId  = blockIdx.x >> 6;
  const int q0   = (blockIdx.x & 63) * 16;
  const float* AoB = Ao_g + bId * NO_ * 64;
  const __bf16* VB = V_g + bId * NO_ * 64;

  // ---- per-lane constant slices -------------------------------------------
  // bf16 A-operand (16x32): element e holds K = e + 8*half + 8*(e>=8)
  // bf16 B-operand (32x16): element e holds K = e + 8*half + 8*(e>=8), N = lq
  float aqv[32], wdv[32];
  v16bf w2b[2];
  {
    const float* aqr = Aq_g + (bId * NQ_ + q0 + lq) * 64;
#pragma unroll
    for (int g2 = 0; g2 < 2; ++g2) {
#pragma unroll
      for (int e = 0; e < 16; ++e) {
        int dA = 32 * g2 + e + 8 * half + ((e >= 8) ? 8 : 0);
        aqv[g2 * 16 + e] = aqr[dA];
        wdv[g2 * 16 + e] = W1[9 * 64 + dA];          // dist row of W1
        int dB = 32 * g2 + e + 8 * half + ((e >= 8) ? 8 : 0);
        w2b[g2][e] = (__bf16)((lq < H_) ? W2[dB * H_ + lq] : 0.0f);
      }
    }
  }
  const float b2v = b2[lq & (H_ - 1)];

  if (lane < 16) {
    const float* p = pos_q_g + (bId * NQ_ + q0 + lane) * 3;
    sQpos[lane][0] = p[0]; sQpos[lane][1] = p[1]; sQpos[lane][2] = p[2];
  }

  v8f O[H_];
#pragma unroll
  for (int h = 0; h < H_; ++h)
#pragma unroll
    for (int r = 0; r < 8; ++r) O[h][r] = 0.0f;

  float runM[H_], runS[H_];
#pragma unroll
  for (int h = 0; h < H_; ++h) { runM[h] = -INFINITY; runS[h] = 0.0f; }

#if GANO_HAVE_TDM
  const i32x8 g1Ao = gano_tdm_g1(2u);        // 4-byte elements
  const i32x8 g1V  = gano_tdm_g1(1u);        // 2-byte elements
  unsigned ldsAo[2], ldsV[2];
#pragma unroll
  for (int p = 0; p < 2; ++p) {
    ldsAo[p] = (unsigned)(unsigned long long)(reinterpret_cast<uintptr_t>(&sAo[p][0]));
    ldsV[p]  = (unsigned)(unsigned long long)(reinterpret_cast<uintptr_t>(&sV[p][0]));
  }
  // Prologue: kick off chunk 0 DMA.
  gano_tdm_load(ldsAo[0], AoB, g1Ao);
  gano_tdm_load(ldsV[0],  VB,  g1V);
#endif

  __syncthreads();

  for (int oc = 0; oc < NCHUNK_; ++oc) {
    const int cur = oc & 1;
    const int o0 = oc * CH_;

#if GANO_HAVE_TDM
    if (oc + 1 < NCHUNK_) {                  // overlap next chunk's DMA
      int nxt = (oc + 1) & 1;
      gano_tdm_load(ldsAo[nxt], AoB + (o0 + CH_) * 64, g1Ao);
      gano_tdm_load(ldsV[nxt],  VB  + (o0 + CH_) * 64, g1V);
    }
    if (oc + 2 < NCHUNK_)                    // warm L2 one chunk further out
      __builtin_prefetch(AoB + (o0 + 2 * CH_) * 64 + lane * 64, 0, 1);
#else
    {   // fallback staging for toolchains without the TDM builtin
#pragma unroll
      for (int j = 0; j < 16; ++j) {
        int idx = lane * 64 + j * 4;
        *(v4f*)&sAo[cur][idx] = *(const v4f*)&AoB[o0 * 64 + idx];
      }
#pragma unroll
      for (int j = 0; j < 8; ++j) {
        int idx = lane * 64 + j * 8;
        *(v4f*)&sV[cur][idx] = *(const v4f*)&VB[o0 * 64 + idx];
      }
    }
#endif

    // Obs positions + mask for this chunk (one obs per lane).
    {
      const float* p = pos_o_g + (bId * NO_ + o0 + lane) * 3;
      sOpos[lane][0] = p[0]; sOpos[lane][1] = p[1]; sOpos[lane][2] = p[2];
      sMask[lane] = omask_g[bId * NO_ + o0 + lane];
    }
    __syncthreads();

    // 32x16 pairwise distances (16 pairs per lane).
#pragma unroll
    for (int pp = 0; pp < 16; ++pp) {
      int pair = lane + pp * 32;
      int o = pair >> 4, q = pair & 15;
      float dx = sQpos[q][0] - sOpos[o][0];
      float dy = sQpos[q][1] - sOpos[o][1];
      float dz = sQpos[q][2] - sOpos[o][2];
      sDist[o][q] = sqrtf(dx * dx + dy * dy + dz * dz);
    }
    __syncthreads();

#if GANO_HAVE_TDM
    if (oc + 1 < NCHUNK_) __builtin_amdgcn_s_wait_tensorcnt((short)2);
    else                  __builtin_amdgcn_s_wait_tensorcnt((short)0);
#endif

    // Logits: hidden(16q x 64) @ W2(64 x 4pad16), 2 bf16 WMMAs per obs.
    for (int t = 0; t < CH_; ++t) {
      float distm = sDist[t][lq];
      const float* ao = &sAo[cur][t * 64];
      v8f c;
#pragma unroll
      for (int r = 0; r < 8; ++r) c[r] = b2v;
#pragma unroll
      for (int g2 = 0; g2 < 2; ++g2) {
        v16bf a;
#pragma unroll
        for (int e = 0; e < 16; ++e) {
          int dA = 32 * g2 + e + 8 * half + ((e >= 8) ? 8 : 0);
          float hv = ao[dA] + aqv[g2 * 16 + e] + distm * wdv[g2 * 16 + e];
          a[e] = (__bf16)fmaxf(hv, 0.0f);            // relu, bf16 for WMMA
        }
        c = WMMA_BF16(a, w2b[g2], c);
      }
      if (lq < H_) {                                  // C: M=q row, N=head col
#pragma unroll
        for (int r = 0; r < 8; ++r) sLog[lq][t][r + half * 8] = c[r];
      }
    }
    __syncthreads();

    // Online softmax row statistics: lanes 0-15 each own one q.
    if (lane < 16) {
      int q = lane;
      unsigned mbits = 0u;                            // bit t = obs t valid
#pragma unroll
      for (int t = 0; t < CH_; ++t)
        if (!(sDist[t][q] > RADIUS_) && sMask[t] != 0) mbits |= (1u << t);
      for (int h = 0; h < H_; ++h) {
        float mNew = runM[h];
#pragma unroll
        for (int t = 0; t < CH_; ++t)
          if (mbits & (1u << t)) mNew = fmaxf(mNew, sLog[h][t][q]);
        float scale, sum = 0.0f;
        if (mNew == -INFINITY) {
          scale = 1.0f;
#pragma unroll
          for (int t = 0; t < CH_; ++t) sLog[h][t][q] = 0.0f;
        } else {
          scale = __expf(runM[h] - mNew);
#pragma unroll
          for (int t = 0; t < CH_; ++t) {
            float p = (mbits & (1u << t)) ? __expf(sLog[h][t][q] - mNew) : 0.0f;
            sLog[h][t][q] = p;
            sum += p;
          }
        }
        runS[h] = runS[h] * scale + sum;
        runM[h] = mNew;
        sScale[h][q] = scale;
      }
    }
    __syncthreads();

    // Rescale accumulators; P(16q x 32o) @ V_h(32o x 16d): 1 bf16 WMMA / head.
#pragma unroll
    for (int h = 0; h < H_; ++h) {
#pragma unroll
      for (int r = 0; r < 8; ++r) O[h][r] *= sScale[h][r + half * 8];
      v16bf a, vb;
#pragma unroll
      for (int e = 0; e < 16; ++e) {
        int kA = e + 8 * half + ((e >= 8) ? 8 : 0);   // obs idx for A elem e
        int kB = e + 8 * half + ((e >= 8) ? 8 : 0);   // obs idx for B elem e
        a[e]  = (__bf16)sLog[h][kA][lq];
        vb[e] = sV[cur][kB * 64 + h * 16 + lq];
      }
      O[h] = WMMA_BF16(a, vb, O[h]);
    }
    __syncthreads();                                  // LDS reuse next chunk
  }

  if (lane < 16)
#pragma unroll
    for (int h = 0; h < H_; ++h) sS[h][lane] = runS[h];
  __syncthreads();

#pragma unroll
  for (int h = 0; h < H_; ++h)
#pragma unroll
    for (int r = 0; r < 8; ++r) {
      int q = r + half * 8;
      float s = sS[h][q];
      float val = (s > 0.0f) ? O[h][r] / s : 0.0f;
      out[(bId * NQ_ + q0 + q) * 64 + h * 16 + lq] = val;
    }
}

// --------------------------------------------------------------------------
extern "C" void kernel_launch(void* const* d_in, const int* in_sizes, int n_in,
                              void* d_out, int out_size, void* d_ws, size_t ws_size,
                              hipStream_t stream) {
  const float* h_obs     = (const float*)d_in[0];
  const float* pos_obs   = (const float*)d_in[1];
  const float* pos_query = (const float*)d_in[2];
  const int*   obs_mask  = (const int*)  d_in[3];
  const float* W1 = (const float*)d_in[4];
  const float* b1 = (const float*)d_in[5];
  const float* W2 = (const float*)d_in[6];
  const float* b2 = (const float*)d_in[7];
  const float* Wv = (const float*)d_in[8];
  const float* bv = (const float*)d_in[9];
  float* out = (float*)d_out;

  // Workspace: Aq f32 | Ao f32 | V bf16  (512KB + 512KB + 256KB)
  float*  Aq = (float*)d_ws;
  float*  Ao = Aq + B_ * NQ_ * D_;
  __bf16* V  = (__bf16*)(Ao + B_ * NO_ * D_);

  const int n = B_ * NQ_ * D_;
  gano_precompute_rows<<<(n + 255) / 256, 256, 0, stream>>>(
      pos_query, W1, b1, Aq, n, 0, 6, 1.0f, 0);
  gano_precompute_rows<<<(n + 255) / 256, 256, 0, stream>>>(
      pos_obs, W1, b1, Ao, n, 3, 6, -1.0f, 1);
  gano_value_proj<<<(B_ * NO_ / 16) * 4, 32, 0, stream>>>(h_obs, Wv, bv, V);
  gano_attention<<<B_ * (NQ_ / 16), 32, 0, stream>>>(
      pos_query, pos_obs, obs_mask, W1, W2, b2, Aq, Ao, V, out);
}
